// TransformerBlock_88072599372193
// MI455X (gfx1250) — compile-verified, matrix operand order checked by reading, not run on hardware
//
#include <hip/hip_runtime.h>
#include <hip/hip_bf16.h>

// ---------------------------------------------------------------------------
// Transformer block (LN -> QKV -> RoPE -> flash attention -> O proj + resid
//                    -> LN -> W1 -> silu*gate -> W2 + resid)
// for B=4, L=2048, D=1024, H=16, HD=64 on gfx1250 (CDNA5, wave32, WMMA).
// GEMMs: v_wmma_f32_16x16x32_f16, 128x64 block tile, LDS double-buffered,
// fed by the Tensor Data Mover (A tiles) and global_load_async_to_lds_b128
// (B tiles).  Each wave computes 32x64 (8 accumulators) so the inner loop is
// 6 ds_load_b128 fragments per 8 WMMAs.
// ---------------------------------------------------------------------------

typedef _Float16 f16;
typedef __attribute__((ext_vector_type(16))) _Float16 v16h;
typedef __attribute__((ext_vector_type(8)))  _Float16 v8h;
typedef __attribute__((ext_vector_type(8)))  float    v8f;
typedef __attribute__((ext_vector_type(4)))  unsigned int u32x4;
typedef __attribute__((ext_vector_type(8)))  int      i32x8;
typedef __attribute__((ext_vector_type(4)))  int      i32x4;

union V16H { v16h v; v8h h[2]; };

static constexpr int kB  = 4;
static constexpr int kL  = 2048;
static constexpr int kD  = 1024;
static constexpr int kH  = 16;
static constexpr int kHD = 64;
static constexpr int kM  = kB * kL;      // 8192 token rows
static constexpr float kEps = 1e-5f;

#if defined(__has_builtin)
# if __has_builtin(__builtin_amdgcn_tensor_load_to_lds)
#  define HAVE_TDM 1
# endif
#endif
#ifndef HAVE_TDM
# define HAVE_TDM 0
#endif

__device__ __forceinline__ v8f wmma_f16(v16h a, v16h b, v8f c) {
  // (neg_a, A, neg_b, B, c_mod, C, reuse_a, reuse_b)
  return __builtin_amdgcn_wmma_f32_16x16x32_f16(false, a, false, b, (short)0, c,
                                                false, false);
}

// Per-lane async copy of 16 bytes: global -> LDS (ASYNCcnt).
__device__ __forceinline__ void async_copy_b128(unsigned lds_off, const void* g) {
  asm volatile("global_load_async_to_lds_b128 %0, %1, off"
               :: "v"(lds_off), "v"((unsigned long long)(size_t)g)
               : "memory");
}
__device__ __forceinline__ void wait_async0() {
  asm volatile("s_wait_asynccnt 0x0" ::: "memory");
}
__device__ __forceinline__ void wait_ds0() {
  asm volatile("s_wait_dscnt 0x0" ::: "memory");
}

#if HAVE_TDM
// Tensor Data Mover: DMA a (tileRows x 64) f16 tile, row stride Kdim elements,
// from global memory into LDS (TENSORcnt).  2-D descriptor; groups 2/3 zero.
__device__ __forceinline__ void tdm_load_tile(unsigned lds_off, const f16* gsrc,
                                              int tileRows, int rowsAvail,
                                              int Kdim) {
  unsigned long long ga = (unsigned long long)(size_t)gsrc;
  u32x4 g0;
  g0.x = 1u;                                    // count=1, user descriptor
  g0.y = lds_off;                               // lds_addr (bytes)
  g0.z = (unsigned)(ga & 0xFFFFFFFFu);          // global_addr[31:0]
  g0.w = (unsigned)((ga >> 32) & 0x01FFFFFFu)   // global_addr[56:32]
         | (2u << 30);                          // type = 2 ("image")
  i32x8 g1;
  g1[0] = 0x00010000;                                   // data_size = 2 bytes
  g1[1] = (int)(((unsigned)Kdim & 0xFFFFu) << 16);      // tensor_dim0[15:0]
  g1[2] = (int)((((unsigned)Kdim >> 16) & 0xFFFFu)      // tensor_dim0[31:16]
         | (((unsigned)rowsAvail & 0xFFFFu) << 16));    // tensor_dim1[15:0]
  g1[3] = (int)((((unsigned)rowsAvail >> 16) & 0xFFFFu)
         | (64u << 16));                                // tile_dim0 = 64
  g1[4] = tileRows;                                     // tile_dim1
  g1[5] = Kdim;                                         // dim0_stride[31:0]
  g1[6] = 0;
  g1[7] = 0;
  i32x4 zz = {0, 0, 0, 0};
#if defined(__clang_major__) && __clang_major__ >= 23
  i32x8 zz8 = {0, 0, 0, 0, 0, 0, 0, 0};
  __builtin_amdgcn_tensor_load_to_lds(g0, g1, zz, zz, zz8, 0);
#else
  __builtin_amdgcn_tensor_load_to_lds(g0, g1, zz, zz, 0);
#endif
}
#endif

// ---------------------------------------------------------------------------
// Weight convert + transpose: Wt[n*K + k] = (f16) W[k*N + n]
// ---------------------------------------------------------------------------
__global__ void wconv_kernel(const float* __restrict__ W, f16* __restrict__ Wt,
                             int Kdim, int Ndim) {
  size_t idx = (size_t)blockIdx.x * blockDim.x + threadIdx.x;
  if (idx >= (size_t)Kdim * Ndim) return;
  int n = (int)(idx / Kdim);
  int k = (int)(idx % Kdim);
  Wt[idx] = (f16)W[(size_t)k * Ndim + n];
}

// ---------------------------------------------------------------------------
// LayerNorm: per-row mean/var, writes f16 activations.
// ---------------------------------------------------------------------------
__global__ __launch_bounds__(256) void layernorm_kernel(
    const float* __restrict__ X, const float* __restrict__ g,
    const float* __restrict__ b, f16* __restrict__ Y, int Ddim) {
  const int row = blockIdx.x;
  const float* x = X + (size_t)row * Ddim;
  __shared__ float red[256];

  float s = 0.f;
  for (int i = threadIdx.x; i < Ddim; i += 256) s += x[i];
  red[threadIdx.x] = s;
  __syncthreads();
  for (int off = 128; off > 0; off >>= 1) {
    if (threadIdx.x < off) red[threadIdx.x] += red[threadIdx.x + off];
    __syncthreads();
  }
  const float mu = red[0] / Ddim;
  __syncthreads();

  float v = 0.f;
  for (int i = threadIdx.x; i < Ddim; i += 256) {
    float d = x[i] - mu;
    v += d * d;
  }
  red[threadIdx.x] = v;
  __syncthreads();
  for (int off = 128; off > 0; off >>= 1) {
    if (threadIdx.x < off) red[threadIdx.x] += red[threadIdx.x + off];
    __syncthreads();
  }
  const float rstd = rsqrtf(red[0] / Ddim + kEps);

  for (int i = threadIdx.x; i < Ddim; i += 256)
    Y[(size_t)row * Ddim + i] = (f16)((x[i] - mu) * rstd * g[i] + b[i]);
}

// ---------------------------------------------------------------------------
// WMMA GEMM: Y[M,N] = A[M,K](f16) * Bt[N,K](f16, pre-transposed) (+ resid)
// Block = 128 threads = 4 waves.  128(M) x 64(N) block tile, BK=64,
// LDS double-buffered: A tile via TDM, B tile via async-to-LDS copies.
// Each wave owns 32(M) x 64(N): acc[0..3] = rows wave*32..+15,
// acc[4..7] = rows wave*32+16..+31.
// ---------------------------------------------------------------------------
__global__ __launch_bounds__(128) void gemm_wmma_kernel(
    const f16* __restrict__ A, const f16* __restrict__ Bt,
    const float* __restrict__ resid, float* __restrict__ Y,
    int Mdim, int Ndim, int Kdim) {
  __shared__ __attribute__((aligned(16))) f16 Abuf[2][128 * 64];  // 2 x 16 KB
  __shared__ __attribute__((aligned(16))) f16 Bbuf[2][64 * 64];   // 2 x  8 KB

  const int tid  = threadIdx.x;
  const int wave = tid >> 5;
  const int lane = tid & 31;
  const int lo = lane & 15;
  const int hi = lane >> 4;
  const int rowBase = blockIdx.y * 128;
  const int colBase = blockIdx.x * 64;

  // Stage one 128(M) x 64(K) A tile and one 64(N) x 64(K) B tile into LDS.
  auto issue_stage = [&](int buf, int kb) {
#if HAVE_TDM
    if (tid == 0)
      tdm_load_tile((unsigned)(size_t)&Abuf[buf][0],
                    A + (size_t)rowBase * Kdim + kb, 128, Mdim, Kdim);
#else
#pragma unroll
    for (int i = 0; i < 8; ++i) {                  // 1024 chunks / 128 threads
      const int c = tid + i * 128;
      const int r = c >> 3, ch = c & 7;
      async_copy_b128((unsigned)(size_t)&Abuf[buf][r * 64 + ch * 8],
                      A + (size_t)(rowBase + r) * Kdim + kb + ch * 8);
    }
#endif
#pragma unroll
    for (int i = 0; i < 4; ++i) {                  // 512 chunks / 128 threads
      const int c = tid + i * 128;
      const int r = c >> 3, ch = c & 7;
      async_copy_b128((unsigned)(size_t)&Bbuf[buf][r * 64 + ch * 8],
                      Bt + (size_t)(colBase + r) * Kdim + kb + ch * 8);
    }
  };

  v8f zero = {};
  v8f acc[8] = {zero, zero, zero, zero, zero, zero, zero, zero};

  const int NK = Kdim / 64;
  issue_stage(0, 0);

  for (int s = 0; s < NK; ++s) {
    const int buf = s & 1;
#if HAVE_TDM
    __builtin_amdgcn_s_wait_tensorcnt(0);
#endif
    wait_async0();
    __syncthreads();

    if (s + 1 < NK) issue_stage(buf ^ 1, (s + 1) * 64);

#pragma unroll
    for (int kk = 0; kk < 64; kk += 32) {
      // issue ALL fragment loads first so WMMAs can go back-to-back
      const int ar0 = (wave * 32 + lo) * 64 + kk + hi * 8;
      const int ar1 = (wave * 32 + 16 + lo) * 64 + kk + hi * 8;
      V16H af0, af1;
      af0.h[0] = *(const v8h*)&Abuf[buf][ar0];
      af0.h[1] = *(const v8h*)&Abuf[buf][ar0 + 16];
      af1.h[0] = *(const v8h*)&Abuf[buf][ar1];
      af1.h[1] = *(const v8h*)&Abuf[buf][ar1 + 16];
      V16H bf[4];
#pragma unroll
      for (int n0 = 0; n0 < 4; ++n0) {
        const int br = (n0 * 16 + lo) * 64 + kk + hi * 8;
        bf[n0].h[0] = *(const v8h*)&Bbuf[buf][br];
        bf[n0].h[1] = *(const v8h*)&Bbuf[buf][br + 16];
      }
#pragma unroll
      for (int n0 = 0; n0 < 4; ++n0) {
        acc[n0]     = wmma_f16(af0.v, bf[n0].v, acc[n0]);
        acc[4 + n0] = wmma_f16(af1.v, bf[n0].v, acc[4 + n0]);
      }
    }
  }

#pragma unroll
  for (int sub = 0; sub < 2; ++sub) {
#pragma unroll
    for (int r = 0; r < 8; ++r) {
      const int grow = rowBase + wave * 32 + sub * 16 + r + 8 * hi;
#pragma unroll
      for (int n0 = 0; n0 < 4; ++n0) {
        const int col = colBase + n0 * 16 + lo;
        float v = acc[sub * 4 + n0][r];
        if (resid) v += resid[(size_t)grow * Ndim + col];
        Y[(size_t)grow * Ndim + col] = v;
      }
    }
  }
}

// ---------------------------------------------------------------------------
// RoPE + split heads: f32 [B*L, D] -> f16 [B, H, L, HD]
// ---------------------------------------------------------------------------
__global__ void rope_kernel(const float* __restrict__ X,
                            const float* __restrict__ C,
                            const float* __restrict__ S,
                            f16* __restrict__ Out) {
  size_t idx = (size_t)blockIdx.x * blockDim.x + threadIdx.x;  // B*L*D
  const int di = (int)(idx % kD);
  const size_t row = idx / kD;  // b*L + l
  const int b = (int)(row / kL);
  const int l = (int)(row % kL);
  const int h = di / kHD;
  const int d = di % kHD;
  const float xv = X[idx];
  const float other = (d < kHD / 2)
                          ? -X[row * kD + h * kHD + d + kHD / 2]
                          :  X[row * kD + h * kHD + d - kHD / 2];
  const float o = xv * C[idx] + other * S[idx];
  Out[(((size_t)b * kH + h) * kL + l) * kHD + d] = (f16)o;
}

// Split heads + transpose: f32 [B*L, D] -> f16 [B, H, HD, L]
__global__ void vsplit_kernel(const float* __restrict__ X, f16* __restrict__ Out) {
  size_t idx = (size_t)blockIdx.x * blockDim.x + threadIdx.x;  // B*L*D
  const int di = (int)(idx % kD);
  const size_t row = idx / kD;
  const int b = (int)(row / kL);
  const int l = (int)(row % kL);
  const int h = di / kHD;
  const int d = di % kHD;
  Out[(((size_t)b * kH + h) * kHD + d) * kL + l] = (f16)X[idx];
}

// ---------------------------------------------------------------------------
// Flash attention: one wave per 16-query tile per (b,h).
// Q/K : f16 [B,H,L,HD]   V : f16 [B,H,HD,L]   ctx out : f16 [B,L,D]
// Online softmax over 32-key steps; S/P round-trip through LDS.
// ---------------------------------------------------------------------------
__global__ __launch_bounds__(32) void flash_attn_kernel(
    const f16* __restrict__ Q, const f16* __restrict__ Kh,
    const f16* __restrict__ Vt, const float* __restrict__ mask,
    f16* __restrict__ ctx) {
  const int bh = blockIdx.y;  // b*H + h
  const int b = bh / kH;
  const int h = bh % kH;
  const int m0 = blockIdx.x * 16;
  const int lane = threadIdx.x;
  const int lo = lane & 15;
  const int hi = lane >> 4;

  __shared__ __attribute__((aligned(16))) float    sS[16 * 32];
  __shared__ __attribute__((aligned(16))) _Float16 sP[16 * 32];
  __shared__ float sRow[16];

  // Q fragments: K-dim = HD = 64 -> two 16x32 A-fragments (reused all steps)
  V16H qf0, qf1;
  const f16* qbase = Q + ((size_t)bh * kL + (m0 + lo)) * kHD;
  qf0.h[0] = *(const v8h*)(qbase + hi * 8);
  qf0.h[1] = *(const v8h*)(qbase + 16 + hi * 8);
  qf1.h[0] = *(const v8h*)(qbase + 32 + hi * 8);
  qf1.h[1] = *(const v8h*)(qbase + 48 + hi * 8);

  v8f zero = {};
  v8f acc[4] = {zero, zero, zero, zero};
  float m_i = -1e30f;
  float l_i = 0.f;
  const float sc = 0.125f;  // 1/sqrt(64)

  for (int j0 = 0; j0 < kL; j0 += 32) {
    // prefetch the next key block
    __builtin_prefetch(Kh + ((size_t)bh * kL + j0 + 32) * kHD, 0, 1);

    // ---- scores S = (Q Kt) * sc + mask, two 16x16 tiles -> LDS -------------
#pragma unroll
    for (int t = 0; t < 2; ++t) {
      const int key = j0 + t * 16 + lo;
      const f16* kb = Kh + ((size_t)bh * kL + key) * kHD;
      V16H k0, k1;
      k0.h[0] = *(const v8h*)(kb + hi * 8);
      k0.h[1] = *(const v8h*)(kb + 16 + hi * 8);
      k1.h[0] = *(const v8h*)(kb + 32 + hi * 8);
      k1.h[1] = *(const v8h*)(kb + 48 + hi * 8);
      v8f s = {};
      s = wmma_f16(qf0.v, k0.v, s);
      s = wmma_f16(qf1.v, k1.v, s);
      const float mval = mask[(size_t)b * kL + key];
#pragma unroll
      for (int r = 0; r < 8; ++r)
        sS[(r + 8 * hi) * 32 + t * 16 + lo] = s[r] * sc + mval;
    }
    wait_ds0();

    // ---- online softmax row update (lanes 0..15 own one row each) ----------
    if (lane < 16) {
      float mx = m_i;
      for (int c = 0; c < 32; ++c) mx = fmaxf(mx, sS[lane * 32 + c]);
      const float corr = __expf(m_i - mx);
      float sum = 0.f;
      for (int c = 0; c < 32; ++c) {
        const float p = __expf(sS[lane * 32 + c] - mx);
        sP[lane * 32 + c] = (f16)p;
        sum += p;
      }
      l_i = l_i * corr + sum;
      m_i = mx;
      sRow[lane] = corr;
    }
    wait_ds0();

    // ---- rescale accumulators by per-row correction ------------------------
#pragma unroll
    for (int r = 0; r < 8; ++r) {
      const float corr = sRow[r + 8 * hi];
#pragma unroll
      for (int n0 = 0; n0 < 4; ++n0) acc[n0][r] *= corr;
    }

    // ---- P as 16x32 A-fragment from LDS ------------------------------------
    V16H pf;
    pf.h[0] = *(const v8h*)&sP[lo * 32 + hi * 8];
    pf.h[1] = *(const v8h*)&sP[lo * 32 + 16 + hi * 8];

    // ---- ctx += P @ V  (V transposed: [hd][key] contiguous over keys) ------
#pragma unroll
    for (int n0 = 0; n0 < 4; ++n0) {
      const int hd = n0 * 16 + lo;
      const f16* vb = Vt + ((size_t)bh * kHD + hd) * kL + j0;
      V16H vf;
      vf.h[0] = *(const v8h*)(vb + hi * 8);
      vf.h[1] = *(const v8h*)(vb + 16 + hi * 8);
      acc[n0] = wmma_f16(pf.v, vf.v, acc[n0]);
    }
  }

  // ---- normalize by row sums and merge heads into [B,L,D] ------------------
  if (lane < 16) sRow[lane] = 1.f / l_i;
  wait_ds0();
#pragma unroll
  for (int r = 0; r < 8; ++r) {
    const int qrow = m0 + r + 8 * hi;
    const float inv = sRow[r + 8 * hi];
#pragma unroll
    for (int n0 = 0; n0 < 4; ++n0) {
      const int hd = n0 * 16 + lo;
      ctx[((size_t)b * kL + qrow) * kD + h * kHD + hd] = (f16)(acc[n0][r] * inv);
    }
  }
}

// ---------------------------------------------------------------------------
// SiLU gate: h [M, 2D] -> g [M, D] f16
// ---------------------------------------------------------------------------
__global__ void silugate_kernel(const float* __restrict__ Hb, f16* __restrict__ G) {
  size_t idx = (size_t)blockIdx.x * blockDim.x + threadIdx.x;  // M*D
  const int c = (int)(idx % kD);
  const size_t row = idx / kD;
  const float x1 = Hb[row * (2 * kD) + c];
  const float x2 = Hb[row * (2 * kD) + kD + c];
  G[idx] = (f16)(x1 / (1.f + __expf(-x1)) * x2);
}

// ---------------------------------------------------------------------------
// Launcher
// ---------------------------------------------------------------------------
extern "C" void kernel_launch(void* const* d_in, const int* in_sizes, int n_in,
                              void* d_out, int out_size, void* d_ws, size_t ws_size,
                              hipStream_t stream) {
  (void)in_sizes; (void)n_in; (void)out_size; (void)ws_size;

  const float* x        = (const float*)d_in[0];
  const float* pos_cos  = (const float*)d_in[1];
  const float* pos_sin  = (const float*)d_in[2];
  const float* attn_mask= (const float*)d_in[3];
  const float* Wq       = (const float*)d_in[4];
  const float* Wk       = (const float*)d_in[5];
  const float* Wv       = (const float*)d_in[6];
  const float* Wo       = (const float*)d_in[7];
  const float* W1       = (const float*)d_in[8];
  const float* W2       = (const float*)d_in[9];
  const float* g1       = (const float*)d_in[10];
  const float* b1       = (const float*)d_in[11];
  const float* g2       = (const float*)d_in[12];
  const float* b2       = (const float*)d_in[13];
  float* out            = (float*)d_out;

  // ---- workspace layout ----------------------------------------------------
  char* ws = (char*)d_ws;
  size_t off = 0;
  auto alloc = [&](size_t bytes) {
    size_t r = off;
    off += (bytes + 255) & ~(size_t)255;
    return r;
  };
  const size_t szW  = (size_t)kD * kD * sizeof(f16);       // 2 MB
  const size_t szW1 = (size_t)kD * 2 * kD * sizeof(f16);   // 4 MB
  const size_t szAh = (size_t)kM * kD * sizeof(f16);       // 16 MB
  const size_t szAf = (size_t)kM * kD * sizeof(float);     // 32 MB

  f16*   WqT  = (f16*)(ws + alloc(szW));
  f16*   WkT  = (f16*)(ws + alloc(szW));
  f16*   WvT  = (f16*)(ws + alloc(szW));
  f16*   WoT  = (f16*)(ws + alloc(szW));
  f16*   W1T  = (f16*)(ws + alloc(szW1));
  f16*   W2T  = (f16*)(ws + alloc(szW));
  f16*   xh1  = (f16*)(ws + alloc(szAh));
  float* qf   = (float*)(ws + alloc(szAf));
  float* kf   = (float*)(ws + alloc(szAf));
  float* vf   = (float*)(ws + alloc(szAf));
  f16*   qh   = (f16*)(ws + alloc(szAh));
  f16*   kh   = (f16*)(ws + alloc(szAh));
  f16*   vt   = (f16*)(ws + alloc(szAh));
  f16*   ctxh = (f16*)(ws + alloc(szAh));
  float* x2   = (float*)(ws + alloc(szAf));
  f16*   xh2  = (f16*)(ws + alloc(szAh));
  float* hbuf = (float*)(ws + alloc((size_t)kM * 2 * kD * sizeof(float)));
  f16*   gh   = (f16*)(ws + alloc(szAh));

  // ---- 1. convert + transpose weights to f16 -------------------------------
  {
    const int th = 256;
    const size_t nW = (size_t)kD * kD, nW1 = (size_t)kD * 2 * kD;
    wconv_kernel<<<dim3((nW + th - 1) / th), th, 0, stream>>>(Wq, WqT, kD, kD);
    wconv_kernel<<<dim3((nW + th - 1) / th), th, 0, stream>>>(Wk, WkT, kD, kD);
    wconv_kernel<<<dim3((nW + th - 1) / th), th, 0, stream>>>(Wv, WvT, kD, kD);
    wconv_kernel<<<dim3((nW + th - 1) / th), th, 0, stream>>>(Wo, WoT, kD, kD);
    wconv_kernel<<<dim3((nW1 + th - 1) / th), th, 0, stream>>>(W1, W1T, kD, 2 * kD);
    wconv_kernel<<<dim3((nW + th - 1) / th), th, 0, stream>>>(W2, W2T, kD, kD);
  }

  // ---- 2. LN1 --------------------------------------------------------------
  layernorm_kernel<<<kM, 256, 0, stream>>>(x, g1, b1, xh1, kD);

  // ---- 3. QKV projections --------------------------------------------------
  {
    dim3 grid(kD / 64, kM / 128);
    gemm_wmma_kernel<<<grid, 128, 0, stream>>>(xh1, WqT, nullptr, qf, kM, kD, kD);
    gemm_wmma_kernel<<<grid, 128, 0, stream>>>(xh1, WkT, nullptr, kf, kM, kD, kD);
    gemm_wmma_kernel<<<grid, 128, 0, stream>>>(xh1, WvT, nullptr, vf, kM, kD, kD);
  }

  // ---- 4. RoPE (q,k) + head-split / transpose (v) --------------------------
  {
    const size_t total = (size_t)kM * kD;
    const int th = 256;
    dim3 grid((total + th - 1) / th);
    rope_kernel<<<grid, th, 0, stream>>>(qf, pos_cos, pos_sin, qh);
    rope_kernel<<<grid, th, 0, stream>>>(kf, pos_cos, pos_sin, kh);
    vsplit_kernel<<<grid, th, 0, stream>>>(vf, vt);
  }

  // ---- 5. flash attention --------------------------------------------------
  flash_attn_kernel<<<dim3(kL / 16, kB * kH), 32, 0, stream>>>(qh, kh, vt,
                                                               attn_mask, ctxh);

  // ---- 6. O projection + residual -----------------------------------------
  gemm_wmma_kernel<<<dim3(kD / 64, kM / 128), 128, 0, stream>>>(ctxh, WoT, x, x2,
                                                                kM, kD, kD);

  // ---- 7. LN2 --------------------------------------------------------------
  layernorm_kernel<<<kM, 256, 0, stream>>>(x2, g2, b2, xh2, kD);

  // ---- 8. FFN up (N = 2D) --------------------------------------------------
  gemm_wmma_kernel<<<dim3(2 * kD / 64, kM / 128), 128, 0, stream>>>(
      xh2, W1T, nullptr, hbuf, kM, 2 * kD, kD);

  // ---- 9. silu gate --------------------------------------------------------
  {
    const size_t total = (size_t)kM * kD;
    silugate_kernel<<<dim3((total + 255) / 256), 256, 0, stream>>>(hbuf, gh);
  }

  // ---- 10. FFN down + residual -> d_out ------------------------------------
  gemm_wmma_kernel<<<dim3(kD / 64, kM / 128), 128, 0, stream>>>(gh, W2T, x2, out,
                                                                kM, kD, kD);
}
